// SparseMLP_35983236006082
// MI455X (gfx1250) — matrix operand order repaired and checked by reading, not compiled, not run on hardware
//
#include <hip/hip_runtime.h>
#include <stdint.h>

// ---------------- problem constants ----------------
#define T_TOK   2048      // B*S tokens
#define HID     1024      // hidden
#define INTER   2048      // intermediate
#define I2      4096      // 2*INTER (interleaved gate/up)
#define NEXP    8
#define MT      32        // token rows per expert tile (2 WMMA M-tiles)
#define BLK     512       // 16 waves

typedef __attribute__((ext_vector_type(16))) __bf16 bf16x16;
typedef __attribute__((ext_vector_type(8)))  __bf16 bf16x8;
typedef __attribute__((ext_vector_type(8)))  float  f32x8;
typedef __attribute__((ext_vector_type(4)))  float  f32x4;

// concat two 8-wide bf16 vectors into the 16-wide WMMA operand
static __device__ __forceinline__ bf16x16 bfcat(bf16x8 lo, bf16x8 hi) {
    return __builtin_shufflevector(lo, hi, 0,1,2,3,4,5,6,7,8,9,10,11,12,13,14,15);
}

// A-operand fragment from LDS (16-bit A 16x32 layout):
// lane L: row M = L%16; K elements {base..base+7} and {base+16..base+23},
// base = k0 + (L/16)*8.  p must point at element [row][base]; 16B aligned.
static __device__ __forceinline__ bf16x16 load_a_lds(const __bf16* p) {
    bf16x8 lo = *(const bf16x8*)(p);
    bf16x8 hi = *(const bf16x8*)(p + 16);
    return bfcat(lo, hi);
}

// B-operand fragment from global fp32 (16-bit B 32x16 layout):
// lane L: N = n0 + L%16; K elements {base..base+7} and {base+16..base+23},
// base = k0 + (L/16)*8. p points at fp32 row [base] col N; ld = row stride.
static __device__ __forceinline__ bf16x16 load_b_gl(const float* p, int ld) {
    bf16x16 r;
#pragma unroll
    for (int i = 0; i < 8; ++i) {
        r[i]     = (__bf16)p[(size_t)i * ld];
        r[8 + i] = (__bf16)p[(size_t)(16 + i) * ld];
    }
    return r;
}

// Async copy of 16 bytes global -> LDS (CDNA5 GLOBAL_LOAD_ASYNC_TO_LDS_B128,
// GVS mode: SGPR base + 32-bit VGPR offset; tracked by ASYNCcnt).
static __device__ __forceinline__ void async_copy16(void* lds_dst,
                                                    const float* sbase,
                                                    unsigned voff_bytes) {
    unsigned lds_addr = (unsigned)(uintptr_t)lds_dst;   // low 32b = LDS byte offset
    asm volatile("global_load_async_to_lds_b128 %0, %1, %2"
                 :: "v"(lds_addr), "v"(voff_bytes), "s"(sbase)
                 : "memory");
}
static __device__ __forceinline__ void async_wait0() {
    asm volatile("s_wait_asynccnt 0" ::: "memory");
}

// ---------------- kernel 1: router ----------------
__global__ void moe_router_kernel(const float* __restrict__ x,
                                  const float* __restrict__ wr,
                                  const float* __restrict__ br,
                                  float* __restrict__ scores,
                                  int* __restrict__ tk0, int* __restrict__ tk1,
                                  float* __restrict__ pw0, float* __restrict__ pw1) {
    int t = blockIdx.x * blockDim.x + threadIdx.x;
    if (t >= T_TOK) return;
    float acc[NEXP];
#pragma unroll
    for (int e = 0; e < NEXP; ++e) acc[e] = br[e];
    const float* xr = x + (size_t)t * HID;
    for (int h = 0; h < HID; ++h) {
        float xv = xr[h];
#pragma unroll
        for (int e = 0; e < NEXP; ++e) acc[e] += xv * wr[h * NEXP + e];
    }
    // top-1 (ties -> lowest index, matching jax top_k)
    int e0 = 0;
#pragma unroll
    for (int e = 1; e < NEXP; ++e) if (acc[e] > acc[e0]) e0 = e;
    // top-2 excluding e0
    int e1 = (e0 == 0) ? 1 : 0;
#pragma unroll
    for (int e = 0; e < NEXP; ++e) if (e != e0 && acc[e] > acc[e1]) e1 = e;
    float d  = acc[e1] - acc[e0];          // <= 0, numerically stable
    float ex = __expf(d);
    float p1 = ex / (1.0f + ex);
    float p0 = 1.0f - p1;
#pragma unroll
    for (int e = 0; e < NEXP; ++e) scores[t * NEXP + e] = 0.0f;
    scores[t * NEXP + e0] = p0;
    scores[t * NEXP + e1] = p1;
    tk0[t] = e0; tk1[t] = e1; pw0[t] = p0; pw1[t] = p1;
}

// ---------------- kernel 2: stable per-expert binning (wave32 ballot) ------
__global__ void moe_bin_kernel(const int* __restrict__ tk0, const int* __restrict__ tk1,
                               const float* __restrict__ pw0, const float* __restrict__ pw1,
                               int* __restrict__ elist_t, float* __restrict__ elist_w,
                               int* __restrict__ counts) {
    int wave = threadIdx.x >> 5;
    int lane = threadIdx.x & 31;
    if (wave >= NEXP) return;
    int e = wave;
    int cnt = 0;
    unsigned lt_mask = (lane == 0) ? 0u : ((1u << lane) - 1u);
    for (int base = 0; base < T_TOK; base += 32) {
        int t = base + lane;
        int h0 = (tk0[t] == e);
        int h1 = (tk1[t] == e);
        int hit = h0 | h1;
        unsigned mask = (unsigned)__ballot(hit);   // wave32: low 32 bits
        int pos = cnt + __popc(mask & lt_mask);
        if (hit) {
            elist_t[e * T_TOK + pos] = t;
            elist_w[e * T_TOK + pos] = h0 ? pw0[t] : pw1[t];
        }
        cnt += __popc(mask);
    }
    if (lane == 0) counts[e] = cnt;
}

// ---------------- kernel 3: fused expert tile (bf16 WMMA, M=32) ------------
__global__ __launch_bounds__(BLK)
void moe_expert_kernel(const float* __restrict__ x,
                       const float* __restrict__ gup,  const float* __restrict__ gupb,
                       const float* __restrict__ dwn,  const float* __restrict__ dwnb,
                       const int* __restrict__ elist_t, const float* __restrict__ elist_w,
                       const int* __restrict__ counts,
                       float* __restrict__ out) {
    int e   = blockIdx.y;
    int cnt = counts[e];
    int m0  = blockIdx.x * MT;
    if (m0 >= cnt) return;

    // Row pad +8 bf16 (16B): row stride 2064B = 516 dwords == 4 mod 64 banks,
    // so the 16 lanes of an A-fragment ds_load_b128 land on distinct banks.
    __shared__ __bf16 Xs[MT][HID + 8];   // ~66 KB: token rows, bf16
    __shared__ __bf16 As[MT][264];       // ~16.9 KB: activated chunk [32 x 256] (+pad)
    __shared__ float  Bg[I2];            // 16 KB: gate_up bias (async-loaded)
    __shared__ float  Bd[HID];           // 4 KB:  down bias    (async-loaded)
    __shared__ int    toks[MT];
    __shared__ float  wgts[MT];

    int tid  = threadIdx.x;
    int wave = tid >> 5;      // 0..15
    int lane = tid & 31;
    int half = lane >> 4;
    int l16  = lane & 15;

    // ---- kick off async bias loads (overlap with X staging) ----
    {
        const float* gb = gupb + (size_t)e * I2;
        const float* db = dwnb + (size_t)e * HID;
        for (int i = tid; i < I2 / 4; i += BLK)         // 2 iters/thread
            async_copy16(&Bg[i * 4], gb, (unsigned)(i * 16));
        for (int i = tid; i < HID / 4; i += BLK)        // waves 0..7 only
            async_copy16(&Bd[i * 4], db, (unsigned)(i * 16));
    }

    if (tid < MT) {
        int idx  = m0 + tid;
        bool val = idx < cnt;
        toks[tid] = val ? elist_t[e * T_TOK + idx] : -1;
        wgts[tid] = val ? elist_w[e * T_TOK + idx] : 0.0f;
    }
    __syncthreads();
    for (int i = tid; i < MT * HID; i += BLK) {
        int r = i >> 10, h = i & (HID - 1);
        int t = toks[r];
        Xs[r][h] = (__bf16)((t >= 0) ? x[(size_t)t * HID + h] : 0.0f);
    }
    async_wait0();            // bias bytes landed in LDS (per issuing wave)
    __syncthreads();

    const float* Wg = gup + (size_t)e * HID * I2;
    const float* Wd = dwn + (size_t)e * INTER * HID;

    f32x8 o[2][4];                       // out tiles: 2 M x 4 N (64 cols/wave)
#pragma unroll
    for (int mt = 0; mt < 2; ++mt)
#pragma unroll
        for (int nt = 0; nt < 4; ++nt)
#pragma unroll
            for (int v = 0; v < 8; ++v) o[mt][nt][v] = 0.0f;

    for (int c = 0; c < 8; ++c) {                 // 8 chunks of 512 gate_up cols
        int nb = c * 512 + wave * 32;             // this wave's 32-column slice
        f32x8 acc[2][2];
#pragma unroll
        for (int mt = 0; mt < 2; ++mt)
#pragma unroll
            for (int nt = 0; nt < 2; ++nt)
#pragma unroll
                for (int v = 0; v < 8; ++v) acc[mt][nt][v] = 0.0f;

        // ---- GEMM1: C[32, 32] += X[32, H] * Wg[H, 32-slice] ----
        for (int k0 = 0; k0 < HID; k0 += 32) {
            bf16x16 a0 = load_a_lds(&Xs[l16][k0 + half * 8]);
            bf16x16 a1 = load_a_lds(&Xs[16 + l16][k0 + half * 8]);
            const float* brow = Wg + (size_t)(k0 + half * 8) * I2 + nb + l16;
            if (k0 + 32 < HID)
                __builtin_prefetch(brow + (size_t)32 * I2, 0, 1);   // global_prefetch
#pragma unroll
            for (int nt = 0; nt < 2; ++nt) {
                bf16x16 bfr = load_b_gl(brow + nt * 16, I2);        // reused by 2 M-tiles
                acc[0][nt] = __builtin_amdgcn_wmma_f32_16x16x32_bf16(
                    false, a0, false, bfr, (short)0, acc[0][nt], false, false);
                acc[1][nt] = __builtin_amdgcn_wmma_f32_16x16x32_bf16(
                    false, a1, false, bfr, (short)0, acc[1][nt], false, false);
            }
        }

        // ---- bias + clamped-SiLU (interleaved gate/up pairing via shfl) ----
#pragma unroll
        for (int nt = 0; nt < 2; ++nt) {
            int ncol   = nb + nt * 16 + l16;         // global gate_up column
            float bias = Bg[ncol];                   // from async-loaded LDS
#pragma unroll
            for (int mt = 0; mt < 2; ++mt) {
#pragma unroll
                for (int v = 0; v < 8; ++v) {
                    float val  = acc[mt][nt][v] + bias;
                    float part = __shfl_xor(val, 1, 32); // partner col (gate<->up)
                    if ((l16 & 1) == 0) {                // even col = gate lane
                        float g = fminf(val, 7.0f);
                        float u = fminf(fmaxf(part, -7.0f), 7.0f);
                        float s = 1.0f / (1.0f + __expf(-1.702f * g));
                        float a = (u + 1.0f) * (g * s);
                        int M    = mt * 16 + v + half * 8;
                        int acol = wave * 16 + nt * 8 + (l16 >> 1);
                        As[M][acol] = (__bf16)a;
                    }
                }
            }
        }
        __syncthreads();

        // ---- GEMM2 partial: O[32, 64] += A[32, 256] * Wd[256-slice, 64] ----
        for (int k0 = 0; k0 < 256; k0 += 32) {
            bf16x16 a0 = load_a_lds(&As[l16][k0 + half * 8]);
            bf16x16 a1 = load_a_lds(&As[16 + l16][k0 + half * 8]);
            const float* brow = Wd + (size_t)(c * 256 + k0 + half * 8) * HID + wave * 64 + l16;
#pragma unroll
            for (int nt = 0; nt < 4; ++nt) {
                bf16x16 bfr = load_b_gl(brow + nt * 16, HID);       // reused by 2 M-tiles
                o[0][nt] = __builtin_amdgcn_wmma_f32_16x16x32_bf16(
                    false, a0, false, bfr, (short)0, o[0][nt], false, false);
                o[1][nt] = __builtin_amdgcn_wmma_f32_16x16x32_bf16(
                    false, a1, false, bfr, (short)0, o[1][nt], false, false);
            }
        }
        __syncthreads();
    }

    // ---- epilogue: down bias (LDS), router weight, scatter-add ----
#pragma unroll
    for (int mt = 0; mt < 2; ++mt) {
#pragma unroll
        for (int v = 0; v < 8; ++v) {
            int row = mt * 16 + v + half * 8;
            int t   = toks[row];
            float w = wgts[row];
            if (t >= 0) {
#pragma unroll
                for (int nt = 0; nt < 4; ++nt) {
                    int col   = wave * 64 + nt * 16 + l16;
                    float val = (o[mt][nt][v] + Bd[col]) * w;
                    atomicAdd(&out[(size_t)t * HID + col], val);
                }
            }
        }
    }
}

// ---------------- host launch ----------------
extern "C" void kernel_launch(void* const* d_in, const int* in_sizes, int n_in,
                              void* d_out, int out_size, void* d_ws, size_t ws_size,
                              hipStream_t stream) {
    (void)in_sizes; (void)n_in; (void)out_size; (void)ws_size;
    const float* x    = (const float*)d_in[0];
    const float* wr   = (const float*)d_in[1];
    const float* br   = (const float*)d_in[2];
    const float* gup  = (const float*)d_in[3];
    const float* gupb = (const float*)d_in[4];
    const float* dwn  = (const float*)d_in[5];
    const float* dwnb = (const float*)d_in[6];

    float* out    = (float*)d_out;
    float* scores = out + (size_t)T_TOK * HID;   // second return value

    // workspace layout
    char* ws = (char*)d_ws;
    int*   tk0     = (int*)ws;                         ws += T_TOK * sizeof(int);
    int*   tk1     = (int*)ws;                         ws += T_TOK * sizeof(int);
    float* pw0     = (float*)ws;                       ws += T_TOK * sizeof(float);
    float* pw1     = (float*)ws;                       ws += T_TOK * sizeof(float);
    int*   elist_t = (int*)ws;                         ws += NEXP * T_TOK * sizeof(int);
    float* elist_w = (float*)ws;                       ws += NEXP * T_TOK * sizeof(float);
    int*   counts  = (int*)ws;                         ws += NEXP * sizeof(int);

    // zero output region (scatter-add target); scores fully overwritten by router
    hipMemsetAsync(d_out, 0, (size_t)T_TOK * HID * sizeof(float), stream);

    moe_router_kernel<<<T_TOK / 256, 256, 0, stream>>>(x, wr, br, scores,
                                                       tk0, tk1, pw0, pw1);
    moe_bin_kernel<<<1, 256, 0, stream>>>(tk0, tk1, pw0, pw1,
                                          elist_t, elist_w, counts);
    dim3 grid(T_TOK / MT, NEXP);
    moe_expert_kernel<<<grid, BLK, 0, stream>>>(x, gup, gupb, dwn, dwnb,
                                                elist_t, elist_w, counts, out);
}
